// BlockSparseMLP_52432960750071
// MI455X (gfx1250) — compile-verified, matrix operand order checked by reading, not run on hardware
//
#include <hip/hip_runtime.h>
#include <hip/hip_bf16.h>
#include <cstdint>

// ---------------------------------------------------------------------------
// MoE (DeepSeek-v3 routing) for MI455X / gfx1250, wave32 + WMMA bf16 +
// async global->LDS double buffering.
// T=1024 tokens, H=1024 hidden, F=512 ffn, E=16 experts, top-4 of 16.
// ---------------------------------------------------------------------------

#define T_TOK 1024
#define H_DIM 1024
#define F_DIM 512
#define E_NUM 16
#define TOPK  4
#define LPITCH 40   // LDS row pitch in bf16 elements (80B: 16B-aligned, bank-spread)

typedef __attribute__((ext_vector_type(16))) __bf16 v16bf;
typedef __attribute__((ext_vector_type(8)))  __bf16 v8bf;
typedef __attribute__((ext_vector_type(8)))  float  v8f;
typedef __attribute__((ext_vector_type(4)))  int    v4i;
typedef __attribute__((ext_vector_type(2)))  int    v2i;

#if __has_builtin(__builtin_amdgcn_global_load_async_to_lds_b128) && \
    __has_builtin(__builtin_amdgcn_global_load_async_to_lds_b64)
#define ASYNC_LDS 1
#else
#define ASYNC_LDS 0
#endif

typedef __attribute__((address_space(1))) v4i glb_v4i;
typedef __attribute__((address_space(3))) v4i lds_v4i;
typedef __attribute__((address_space(1))) v2i glb_v2i;
typedef __attribute__((address_space(3))) v2i lds_v2i;

// 16-byte global -> LDS copy (async on CDNA5; tracked by ASYNCcnt)
__device__ __forceinline__ void cp16(void* lds, const void* g) {
#if ASYNC_LDS
  __builtin_amdgcn_global_load_async_to_lds_b128(
      (glb_v4i*)(uintptr_t)g,
      (lds_v4i*)(uint32_t)(uintptr_t)lds, 0, 0);
#else
  *(uint4*)lds = *(const uint4*)g;
#endif
}
// 8-byte global -> LDS copy
__device__ __forceinline__ void cp8(void* lds, const void* g) {
#if ASYNC_LDS
  __builtin_amdgcn_global_load_async_to_lds_b64(
      (glb_v2i*)(uintptr_t)g,
      (lds_v2i*)(uint32_t)(uintptr_t)lds, 0, 0);
#else
  *(uint2*)lds = *(const uint2*)g;
#endif
}
__device__ __forceinline__ void wait_async0() {
#if ASYNC_LDS
#if __has_builtin(__builtin_amdgcn_s_wait_asynccnt)
  __builtin_amdgcn_s_wait_asynccnt(0);
#else
  asm volatile("s_wait_asynccnt 0x0" ::: "memory");
#endif
#endif
}

__device__ __forceinline__ float frcp(float x) {
#if __has_builtin(__builtin_amdgcn_rcpf)
  return __builtin_amdgcn_rcpf(x);
#else
  return 1.f / x;
#endif
}
__device__ __forceinline__ float fsigmoid(float x) {
  return frcp(1.f + __expf(-x));
}

__device__ __forceinline__ __bf16 f2bf(float f) {
  union { float f; uint32_t u; } v; v.f = f;
  uint32_t r = v.u + 0x7FFFu + ((v.u >> 16) & 1u);   // round-to-nearest-even
  uint16_t h = (uint16_t)(r >> 16);
  return __builtin_bit_cast(__bf16, h);
}

// Build a 16x32 bf16 A/B fragment per ISA 7.12.2: two 16-byte contiguous
// chunks per lane (lanes 0-15: K 0-7 & 16-23; lanes 16-31: K 8-15 & 24-31).
__device__ __forceinline__ v16bf ld_frag(const __bf16* p0, const __bf16* p1) {
  v8bf lo = *(const v8bf*)p0;
  v8bf hi = *(const v8bf*)p1;
  v16bf r;
  #pragma unroll
  for (int i = 0; i < 8; ++i) { r[i] = lo[i]; r[i + 8] = hi[i]; }
  return r;
}

// ---------------------------------------------------------------------------
// 0) zero output + expert counters (harness poisons buffers with 0xAA)
// ---------------------------------------------------------------------------
__global__ void zero_kernel(float* __restrict__ out, int n, int* __restrict__ counts) {
  int i = blockIdx.x * blockDim.x + threadIdx.x;
  if (i < n) out[i] = 0.f;
  if (blockIdx.x == 0 && threadIdx.x < E_NUM) counts[threadIdx.x] = 0;
}

// ---------------------------------------------------------------------------
// 1) fp32 -> bf16 conversion of x (row major, K-contiguous for WMMA A)
// ---------------------------------------------------------------------------
__global__ void convx_kernel(const float* __restrict__ in, __bf16* __restrict__ out, int n) {
  int i = (blockIdx.x * blockDim.x + threadIdx.x) * 4;
  if (i < n) {
    float4 v = *(const float4*)(in + i);
    out[i + 0] = f2bf(v.x); out[i + 1] = f2bf(v.y);
    out[i + 2] = f2bf(v.z); out[i + 3] = f2bf(v.w);
  }
}

// ---------------------------------------------------------------------------
// 2) per-expert transpose+convert: in [e][R][C] fp32 -> out [e][C][R] bf16.
//    Makes every WMMA B-operand contiguous along K.
// ---------------------------------------------------------------------------
__global__ void tconv_kernel(const float* __restrict__ in, __bf16* __restrict__ out,
                             int R, int C) {
  __shared__ float tile[32][33];
  int e = blockIdx.z;
  const float* src = in + (size_t)e * R * C;
  __bf16* dst = out + (size_t)e * R * C;
  int c0 = blockIdx.x * 32, r0 = blockIdx.y * 32;
  int tx = threadIdx.x, ty = threadIdx.y;                 // 32 x 8
  #pragma unroll
  for (int i = 0; i < 32; i += 8)
    tile[ty + i][tx] = src[(size_t)(r0 + ty + i) * C + (c0 + tx)];
  __syncthreads();
  #pragma unroll
  for (int i = 0; i < 32; i += 8)
    dst[(size_t)(c0 + ty + i) * R + (r0 + tx)] = f2bf(tile[tx][ty + i]);
}

// ---------------------------------------------------------------------------
// 3) routing: one wave32 per token. Lane-parallel 16-expert GEMV over H,
//    butterfly reduce, DS3 group-limited top-4 in lane 0, atomic compaction.
// ---------------------------------------------------------------------------
__global__ __launch_bounds__(256)
void route_kernel(const float* __restrict__ x, const float* __restrict__ gate,
                  const float* __restrict__ ebias,
                  int* __restrict__ counts, int* __restrict__ tok_list,
                  float* __restrict__ w_list) {
  const int lane = threadIdx.x & 31;
  const int wv   = threadIdx.x >> 5;
  const int t    = blockIdx.x * 8 + wv;          // T_TOK == gridDim.x * 8

  float acc[E_NUM];
  #pragma unroll
  for (int e = 0; e < E_NUM; ++e) acc[e] = 0.f;

  const float* xr = x + (size_t)t * H_DIM;
  for (int h = lane; h < H_DIM; h += 32) {
    float xv = xr[h];
    const float* gr = gate + (size_t)h * E_NUM;
    #pragma unroll
    for (int e = 0; e < E_NUM; ++e) acc[e] += xv * gr[e];
  }
  #pragma unroll
  for (int off = 16; off > 0; off >>= 1) {
    #pragma unroll
    for (int e = 0; e < E_NUM; ++e) acc[e] += __shfl_xor(acc[e], off, 32);
  }

  if (lane == 0) {
    float sc[E_NUM], sfc[E_NUM];
    #pragma unroll
    for (int e = 0; e < E_NUM; ++e) {
      sc[e]  = fsigmoid(acc[e]);                 // sigmoid scores
      sfc[e] = sc[e] + ebias[e];                 // scores_for_choice
    }
    float gs[4];
    #pragma unroll
    for (int g = 0; g < 4; ++g) {
      float m1 = -1e30f, m2 = -1e30f;
      #pragma unroll
      for (int j = 0; j < 4; ++j) {
        float v = sfc[g * 4 + j];
        if (v > m1) { m2 = m1; m1 = v; } else if (v > m2) { m2 = v; }
      }
      gs[g] = m1 + m2;
    }
    int g0 = 0;
    for (int g = 1; g < 4; ++g) if (gs[g] > gs[g0]) g0 = g;
    gs[g0] = -1e30f;
    int g1 = 0;
    for (int g = 1; g < 4; ++g) if (gs[g] > gs[g1]) g1 = g;

    float masked[E_NUM];
    #pragma unroll
    for (int e = 0; e < E_NUM; ++e) {
      int g = e >> 2;
      masked[e] = (g == g0 || g == g1) ? sfc[e] : 0.f;
    }
    int   idx[TOPK];
    float w[TOPK];
    float wsum = 0.f;
    #pragma unroll
    for (int k = 0; k < TOPK; ++k) {
      int j = 0;
      for (int e = 1; e < E_NUM; ++e) if (masked[e] > masked[j]) j = e;
      masked[j] = -1e30f;
      idx[k] = j;
      w[k] = sc[j];
      wsum += w[k];
    }
    float scale = 2.5f * frcp(wsum + 1e-20f);    // ROUTED_SCALING_FACTOR
    #pragma unroll
    for (int k = 0; k < TOPK; ++k) {
      int e = idx[k];
      int pos = atomicAdd(&counts[e], 1);
      tok_list[(size_t)e * T_TOK + pos] = t;
      w_list [(size_t)e * T_TOK + pos] = w[k] * scale;
    }
  }
}

// ---------------------------------------------------------------------------
// 4) gate/up GEMM + SiLU*mul, per expert over its compacted token list.
//    Block tile: 32 tokens x 128 F-cols, K = H = 1024 in 32-chunks,
//    double-buffered async global->LDS staging overlapped with WMMA.
//    8 waves in 2(M) x 4(N): each wave 16 rows x 2 n-tiles, gate AND up.
// ---------------------------------------------------------------------------
__global__ __launch_bounds__(256)
void gateup_kernel(const __bf16* __restrict__ xb,
                   const __bf16* __restrict__ wg,   // [E][F][H] (transposed)
                   const __bf16* __restrict__ wu,   // [E][F][H]
                   const int* __restrict__ counts,
                   const int* __restrict__ tok_list,
                   __bf16* __restrict__ act) {      // [E][T][F]
  const int e  = blockIdx.z;
  const int m0 = blockIdx.y * 32;
  const int n0 = blockIdx.x * 128;
  const int cnt = counts[e];
  if (m0 >= cnt) return;

  __shared__ __attribute__((aligned(16))) __bf16 As[2][32][LPITCH];
  __shared__ __attribute__((aligned(16))) __bf16 Bg[2][128][LPITCH];
  __shared__ __attribute__((aligned(16))) __bf16 Bu[2][128][LPITCH];
  __shared__ int toks[32];

  const int tid = threadIdx.x;
  if (tid < 32) {
    int s = m0 + tid;
    toks[tid] = tok_list[(size_t)e * T_TOK + (s < cnt ? s : cnt - 1)];
  }
  __syncthreads();

  const int lane = tid & 31;
  const int wid  = tid >> 5;
  const int wm   = (wid >> 2) * 16;     // 0 or 16
  const int wn   = (wid & 3) * 32;      // 0,32,64,96
  const int fr   = lane & 15;           // fragment row
  const int fc   = (lane >> 4) * 8;     // fragment k-chunk base

  const __bf16* wgb = wg + ((size_t)e * F_DIM + n0) * H_DIM;
  const __bf16* wub = wu + ((size_t)e * F_DIM + n0) * H_DIM;

  // stage one 32-wide K chunk into LDS buffer `buf` (async)
  auto stage = [&](int buf, int k0) {
    {
      int r = tid >> 3, c = (tid & 7) * 4;                 // 32 rows x 8x4-elem
      cp8(&As[buf][r][c], xb + (size_t)toks[r] * H_DIM + k0 + c);
    }
    {
      int r = tid >> 1, c = (tid & 1) * 16;                // 128 rows x 2x16-elem
      const __bf16* sg = wgb + (size_t)r * H_DIM + k0 + c;
      const __bf16* su = wub + (size_t)r * H_DIM + k0 + c;
      cp16(&Bg[buf][r][c],     sg);
      cp16(&Bg[buf][r][c + 8], sg + 8);
      cp16(&Bu[buf][r][c],     su);
      cp16(&Bu[buf][r][c + 8], su + 8);
    }
  };

  v8f accg[2], accu[2];
  #pragma unroll
  for (int i = 0; i < 2; ++i) {
    #pragma unroll
    for (int j = 0; j < 8; ++j) { accg[i][j] = 0.f; accu[i][j] = 0.f; }
  }

  stage(0, 0);
  for (int k0 = 0; k0 < H_DIM; k0 += 32) {
    const int cur = (k0 >> 5) & 1;
    wait_async0();          // our async writes to buffer `cur` are done
    __syncthreads();        // everyone's are done; prior compute finished
    if (k0 + 32 < H_DIM) stage(cur ^ 1, k0 + 32);   // prefetch next chunk

    v16bf a = ld_frag(&As[cur][wm + fr][fc], &As[cur][wm + fr][fc + 16]);
    #pragma unroll
    for (int nt = 0; nt < 2; ++nt) {
      const int br = wn + nt * 16 + fr;
      v16bf bg = ld_frag(&Bg[cur][br][fc], &Bg[cur][br][fc + 16]);
      v16bf bu = ld_frag(&Bu[cur][br][fc], &Bu[cur][br][fc + 16]);
      accg[nt] = __builtin_amdgcn_wmma_f32_16x16x32_bf16(
          false, a, false, bg, (short)0, accg[nt], false, false);
      accu[nt] = __builtin_amdgcn_wmma_f32_16x16x32_bf16(
          false, a, false, bu, (short)0, accu[nt], false, false);
    }
  }

  // epilogue: silu(g) * u -> bf16 activations (C/D layout: VGPR r -> M=r / r+8)
  #pragma unroll
  for (int nt = 0; nt < 2; ++nt) {
    #pragma unroll
    for (int r = 0; r < 8; ++r) {
      int mrow = wm + r + (lane >> 4) * 8;
      int col  = n0 + wn + nt * 16 + (lane & 15);
      float g = accg[nt][r], u = accu[nt][r];
      float s = g * fsigmoid(g) * u;
      act[((size_t)e * T_TOK + (m0 + mrow)) * F_DIM + col] = f2bf(s);
    }
  }
}

// ---------------------------------------------------------------------------
// 5) down GEMM + weighted scatter into out via global_atomic_add_f32.
//    Block tile: 32 slots x 128 H-cols, K = F = 512, double-buffered async.
// ---------------------------------------------------------------------------
__global__ __launch_bounds__(256)
void down_kernel(const __bf16* __restrict__ act,   // [E][T][F]
                 const __bf16* __restrict__ wd,    // [E][H][F] (transposed)
                 const int* __restrict__ counts,
                 const int* __restrict__ tok_list,
                 const float* __restrict__ w_list,
                 float* __restrict__ out) {
  const int e  = blockIdx.z;
  const int m0 = blockIdx.y * 32;
  const int n0 = blockIdx.x * 128;
  const int cnt = counts[e];
  if (m0 >= cnt) return;

  __shared__ __attribute__((aligned(16))) __bf16 As[2][32][LPITCH];
  __shared__ __attribute__((aligned(16))) __bf16 Bs[2][128][LPITCH];
  __shared__ int   toks[32];
  __shared__ float wts[32];

  const int tid = threadIdx.x;
  if (tid < 32) {
    int s = m0 + tid;
    int ss = (s < cnt ? s : cnt - 1);
    toks[tid] = tok_list[(size_t)e * T_TOK + ss];
    wts[tid]  = w_list [(size_t)e * T_TOK + ss];
  }
  __syncthreads();

  const int lane = tid & 31;
  const int wid  = tid >> 5;
  const int wm   = (wid >> 2) * 16;
  const int wn   = (wid & 3) * 32;
  const int fr   = lane & 15;
  const int fc   = (lane >> 4) * 8;

  const __bf16* ab  = act + ((size_t)e * T_TOK + m0) * F_DIM;
  const __bf16* wdb = wd + ((size_t)e * H_DIM + n0) * F_DIM;

  auto stage = [&](int buf, int k0) {
    {
      int r = tid >> 3, c = (tid & 7) * 4;
      cp8(&As[buf][r][c], ab + (size_t)r * F_DIM + k0 + c);
    }
    {
      int r = tid >> 1, c = (tid & 1) * 16;
      const __bf16* sb = wdb + (size_t)r * F_DIM + k0 + c;
      cp16(&Bs[buf][r][c],     sb);
      cp16(&Bs[buf][r][c + 8], sb + 8);
    }
  };

  v8f acc[2];
  #pragma unroll
  for (int i = 0; i < 2; ++i) {
    #pragma unroll
    for (int j = 0; j < 8; ++j) acc[i][j] = 0.f;
  }

  stage(0, 0);
  for (int k0 = 0; k0 < F_DIM; k0 += 32) {
    const int cur = (k0 >> 5) & 1;
    wait_async0();
    __syncthreads();
    if (k0 + 32 < F_DIM) stage(cur ^ 1, k0 + 32);

    v16bf a = ld_frag(&As[cur][wm + fr][fc], &As[cur][wm + fr][fc + 16]);
    #pragma unroll
    for (int nt = 0; nt < 2; ++nt) {
      const int br = wn + nt * 16 + fr;
      v16bf b = ld_frag(&Bs[cur][br][fc], &Bs[cur][br][fc + 16]);
      acc[nt] = __builtin_amdgcn_wmma_f32_16x16x32_bf16(
          false, a, false, b, (short)0, acc[nt], false, false);
    }
  }

  #pragma unroll
  for (int nt = 0; nt < 2; ++nt) {
    #pragma unroll
    for (int r = 0; r < 8; ++r) {
      int mrow = wm + r + (lane >> 4) * 8;
      int slot = m0 + mrow;
      if (slot < cnt) {
        int   tok = toks[mrow];
        float w   = wts[mrow];
        int   col = n0 + wn + nt * 16 + (lane & 15);
        atomicAdd(&out[(size_t)tok * H_DIM + col], w * acc[nt][r]);
      }
    }
  }
}

// ---------------------------------------------------------------------------
// Host-side launch (graph-capture safe: only kernel launches on `stream`)
// ---------------------------------------------------------------------------
extern "C" void kernel_launch(void* const* d_in, const int* in_sizes, int n_in,
                              void* d_out, int out_size, void* d_ws, size_t ws_size,
                              hipStream_t stream) {
  const float* x    = (const float*)d_in[0];   // [T,H]
  const float* gate = (const float*)d_in[1];   // [H,E]
  const float* eb   = (const float*)d_in[2];   // [E]
  const float* Wg   = (const float*)d_in[3];   // [E,H,F]
  const float* Wu   = (const float*)d_in[4];   // [E,H,F]
  const float* Wd   = (const float*)d_in[5];   // [E,F,H]
  float* out = (float*)d_out;                  // [T,H]

  char* ws = (char*)d_ws;
  size_t off = 0;
  auto carve = [&](size_t bytes) -> void* {
    void* p = ws + off;
    off += (bytes + 255) & ~(size_t)255;
    return p;
  };
  int*    counts   = (int*)carve(E_NUM * sizeof(int));
  int*    tok_list = (int*)carve((size_t)E_NUM * T_TOK * sizeof(int));
  float*  w_list   = (float*)carve((size_t)E_NUM * T_TOK * sizeof(float));
  __bf16* xb   = (__bf16*)carve((size_t)T_TOK * H_DIM * 2);
  __bf16* wg_t = (__bf16*)carve((size_t)E_NUM * F_DIM * H_DIM * 2);
  __bf16* wu_t = (__bf16*)carve((size_t)E_NUM * F_DIM * H_DIM * 2);
  __bf16* wd_t = (__bf16*)carve((size_t)E_NUM * H_DIM * F_DIM * 2);
  __bf16* act  = (__bf16*)carve((size_t)E_NUM * T_TOK * F_DIM * 2);
  (void)in_sizes; (void)n_in; (void)out_size; (void)ws_size;

  zero_kernel<<<(T_TOK * H_DIM + 255) / 256, 256, 0, stream>>>(out, T_TOK * H_DIM, counts);
  convx_kernel<<<(T_TOK * H_DIM / 4 + 255) / 256, 256, 0, stream>>>(x, xb, T_TOK * H_DIM);

  dim3 tb(32, 8);
  tconv_kernel<<<dim3(F_DIM / 32, H_DIM / 32, E_NUM), tb, 0, stream>>>(Wg, wg_t, H_DIM, F_DIM);
  tconv_kernel<<<dim3(F_DIM / 32, H_DIM / 32, E_NUM), tb, 0, stream>>>(Wu, wu_t, H_DIM, F_DIM);
  tconv_kernel<<<dim3(H_DIM / 32, F_DIM / 32, E_NUM), tb, 0, stream>>>(Wd, wd_t, F_DIM, H_DIM);

  route_kernel<<<T_TOK / 8, 256, 0, stream>>>(x, gate, eb, counts, tok_list, w_list);

  gateup_kernel<<<dim3(F_DIM / 128, T_TOK / 32, E_NUM), 256, 0, stream>>>(
      xb, wg_t, wu_t, counts, tok_list, act);
  down_kernel<<<dim3(H_DIM / 128, T_TOK / 32, E_NUM), 256, 0, stream>>>(
      act, wd_t, counts, tok_list, w_list, out);
}
